// CustomGNN_9345848836441
// MI455X (gfx1250) — matrix operand order repaired and checked by reading, not compiled, not run on hardware
//
#include <hip/hip_runtime.h>
#include <hip/hip_bf16.h>

typedef __attribute__((ext_vector_type(16))) _Float16 v16h;
typedef __attribute__((ext_vector_type(8)))  _Float16 v8h;
typedef __attribute__((ext_vector_type(8)))  float    v8f;

#define Nn   20000
#define Ee   320000
#define E2T  340000
#define FNd  32
#define FEd  16
#define Cc   64
#define Hh   8
#define HC   512
#define NGg  64

__device__ __forceinline__ void atomAddF(float* p, float v) {
  (void)__hip_atomic_fetch_add(p, v, __ATOMIC_RELAXED, __HIP_MEMORY_SCOPE_AGENT);
}
// monotone float -> uint key (order-preserving); memset(0) init == below all reals
__device__ __forceinline__ unsigned fkey(float f) {
  unsigned u = __float_as_uint(f);
  return (u & 0x80000000u) ? ~u : (u | 0x80000000u);
}
__device__ __forceinline__ float funkey(unsigned u) {
  unsigned b = (u & 0x80000000u) ? (u ^ 0x80000000u) : ~u;
  return __uint_as_float(b);
}

// ---------------- self-loop attr mean (fill_value='mean') ----------------
__global__ void loop_sum_kernel(const int* __restrict__ ei, const float* __restrict__ ea,
                                float* __restrict__ lsum, float* __restrict__ lcnt) {
  int id = blockIdx.x * 256 + threadIdx.x;
  if (id >= Ee * FEd) return;
  int e = id >> 4, f = id & 15;
  int col = ei[Ee + e];
  atomAddF(&lsum[col * FEd + f], ea[id]);
  if (f == 0) atomAddF(&lcnt[col], 1.0f);
}

__global__ void build_ea2_kernel(const float* __restrict__ ea, const float* __restrict__ lsum,
                                 const float* __restrict__ lcnt, float* __restrict__ ea2) {
  int id = blockIdx.x * 256 + threadIdx.x;
  if (id >= E2T * FEd) return;
  if (id < Ee * FEd) {
    ea2[id] = ea[id];
  } else {
    int i2 = id - Ee * FEd;
    int n = i2 >> 4, f = i2 & 15;
    ea2[id] = lsum[n * FEd + f] / fmaxf(lcnt[n], 1.0f);
  }
}

// ---------------- weight transpose + f16 convert: W[K][J] -> Wt[J][K] ----
__global__ void conv_transpose_f16(const float* __restrict__ W, _Float16* __restrict__ Wt,
                                   int K, int J) {
  int id = blockIdx.x * 256 + threadIdx.x;
  if (id >= K * J) return;
  int j = id / K, k = id % K;
  Wt[id] = (_Float16)W[k * J + j];
}

// ---------------- bulk f32 -> f16 convert (activations) ----------------
__global__ void conv_f16_kernel(const float* __restrict__ X, _Float16* __restrict__ Xh,
                                long long n) {
  long long id = (long long)blockIdx.x * 256 + threadIdx.x;
  if (id >= n) return;
  Xh[id] = (_Float16)X[id];
}

// we_att[f,h] = sum_c We[f, h*C+c] * a_e[h,c]
__global__ void we_att_kernel(const float* __restrict__ We, const float* __restrict__ Ae,
                              float* __restrict__ weatt) {
  int id = blockIdx.x * 256 + threadIdx.x;
  if (id >= FEd * Hh) return;
  int f = id >> 3, h = id & 7;
  float acc = 0.0f;
  for (int c = 0; c < Cc; ++c) acc += We[f * HC + h * Cc + c] * Ae[h * Cc + c];
  weatt[id] = acc;
}

// ---- WMMA GEMM: Cout[M,Ncols] = A[M,K](f16) * Bt[Ncols,K](f16)^T
// Each wave computes a 16x64 strip (4 n-tiles) so one A fragment feeds 4 WMMAs.
__global__ void gemm_wmma_f16_kernel(const _Float16* __restrict__ A,
                                     const _Float16* __restrict__ Bt,
                                     const float* __restrict__ bias,
                                     float* __restrict__ Cout,
                                     int M, int Ncols, int K, int fuseBiasRelu) {
  const int wave = threadIdx.x >> 5;
  const int lane = threadIdx.x & 31;
  const int ngroups = Ncols >> 6;                   // groups of 4 n-tiles
  const int tile = blockIdx.x * 8 + wave;
  if (tile >= (M >> 4) * ngroups) return;           // wave-uniform: EXEC stays all-1s
  const int mt = tile / ngroups;
  const int ng = tile % ngroups;
  const int half = lane >> 4;
  const int lm = lane & 15;

  v8f c0 = {}, c1 = {}, c2 = {}, c3 = {};
  const _Float16* arow = A + (size_t)(mt * 16 + lm) * K;
  const _Float16* br0 = Bt + (size_t)((ng * 4 + 0) * 16 + lm) * K;
  const _Float16* br1 = Bt + (size_t)((ng * 4 + 1) * 16 + lm) * K;
  const _Float16* br2 = Bt + (size_t)((ng * 4 + 2) * 16 + lm) * K;
  const _Float16* br3 = Bt + (size_t)((ng * 4 + 3) * 16 + lm) * K;

  for (int ks = 0; ks < K; ks += 32) {
    // A fragment: lane half selects K 0-15 vs 16-31, contiguous (matches B)
    const v8h* ap = (const v8h*)(arow + ks + half * 16);
    v8h alo = ap[0], ahi = ap[1];
    v16h a;
#pragma unroll
    for (int i = 0; i < 8; ++i) { a[i] = alo[i]; a[8 + i] = ahi[i]; }
    // B fragments (32x16 col-major rows): lane half selects K 0-15 vs 16-31, contiguous
    const v8h* bp0 = (const v8h*)(br0 + ks + half * 16);
    const v8h* bp1 = (const v8h*)(br1 + ks + half * 16);
    const v8h* bp2 = (const v8h*)(br2 + ks + half * 16);
    const v8h* bp3 = (const v8h*)(br3 + ks + half * 16);
    v16h b0, b1, b2, b3;
    {
      v8h l0 = bp0[0], h0 = bp0[1], l1 = bp1[0], h1 = bp1[1];
      v8h l2 = bp2[0], h2 = bp2[1], l3 = bp3[0], h3 = bp3[1];
#pragma unroll
      for (int i = 0; i < 8; ++i) {
        b0[i] = l0[i]; b0[8 + i] = h0[i];
        b1[i] = l1[i]; b1[8 + i] = h1[i];
        b2[i] = l2[i]; b2[8 + i] = h2[i];
        b3[i] = l3[i]; b3[8 + i] = h3[i];
      }
    }
    c0 = __builtin_amdgcn_wmma_f32_16x16x32_f16(false, a, false, b0, (short)0, c0, false, false);
    c1 = __builtin_amdgcn_wmma_f32_16x16x32_f16(false, a, false, b1, (short)0, c1, false, false);
    c2 = __builtin_amdgcn_wmma_f32_16x16x32_f16(false, a, false, b2, (short)0, c2, false, false);
    c3 = __builtin_amdgcn_wmma_f32_16x16x32_f16(false, a, false, b3, (short)0, c3, false, false);
  }

  // C/D layout: VGPR r -> M = r (+8 for hi lanes), N = lane%16
#pragma unroll
  for (int j = 0; j < 4; ++j) {
    const v8f& c = (j == 0) ? c0 : (j == 1) ? c1 : (j == 2) ? c2 : c3;
    int n = (ng * 4 + j) * 16 + lm;
    float bi = fuseBiasRelu ? bias[n] : 0.0f;
#pragma unroll
    for (int r = 0; r < 8; ++r) {
      int m = mt * 16 + r + half * 8;
      float v = c[r];
      if (fuseBiasRelu) v = fmaxf(v + bi, 0.0f);
      Cout[(size_t)m * Ncols + n] = v;
    }
  }
}

// ---------------- per-node attention scores ----------------
__global__ void scores_kernel(const float* __restrict__ hbuf, const float* __restrict__ As,
                              const float* __restrict__ Ad, float* __restrict__ ssrc,
                              float* __restrict__ sdst) {
  int id = blockIdx.x * 256 + threadIdx.x;
  if (id >= Nn * Hh) return;
  int n = id >> 3, h = id & 7;
  const float* hp = hbuf + (size_t)n * HC + h * Cc;
  float s = 0.0f, d = 0.0f;
  for (int c = 0; c < Cc; ++c) { float v = hp[c]; s += v * As[h * Cc + c]; d += v * Ad[h * Cc + c]; }
  ssrc[id] = s; sdst[id] = d;
}

// ---------------- edge alpha pass 1: raw logits + segment max ----------------
__global__ void alpha1_kernel(const int* __restrict__ ei, const float* __restrict__ ea2,
                              const float* __restrict__ weatt, const float* __restrict__ ssrc,
                              const float* __restrict__ sdst, float* __restrict__ alpha,
                              unsigned* __restrict__ mmax) {
  int id = blockIdx.x * 256 + threadIdx.x;
  if (id >= E2T * Hh) return;
  int e = id >> 3, h = id & 7;
  int r = (e < Ee) ? ei[e] : (e - Ee);
  int cn = (e < Ee) ? ei[Ee + e] : (e - Ee);
  float acc = ssrc[r * Hh + h] + sdst[cn * Hh + h];
  const float* ep = ea2 + (size_t)e * FEd;
#pragma unroll
  for (int f = 0; f < FEd; ++f) acc += ep[f] * weatt[f * Hh + h];
  acc = (acc > 0.0f) ? acc : 0.2f * acc;          // leaky_relu(0.2)
  alpha[id] = acc;
  atomicMax(&mmax[cn * Hh + h], fkey(acc));
}

// ---------------- edge alpha pass 2: exp + segment sum ----------------
__global__ void alpha2_kernel(const int* __restrict__ ei, float* __restrict__ alpha,
                              const unsigned* __restrict__ mmax, float* __restrict__ denom) {
  int id = blockIdx.x * 256 + threadIdx.x;
  if (id >= E2T * Hh) return;
  int e = id >> 3, h = id & 7;
  int cn = (e < Ee) ? ei[Ee + e] : (e - Ee);
  float m = funkey(mmax[cn * Hh + h]);
  float v = expf(alpha[id] - m);
  alpha[id] = v;
  atomAddF(&denom[cn * Hh + h], v);
}

// ---------------- message aggregation: one wave per edge ----------------
__global__ void agg_kernel(const int* __restrict__ ei, const float* __restrict__ hbuf,
                           const float* __restrict__ alpha, const float* __restrict__ denom,
                           float* __restrict__ agg) {
  int e = blockIdx.x * 8 + (threadIdx.x >> 5);
  if (e >= E2T) return;
  int lane = threadIdx.x & 31;
  int r = (e < Ee) ? ei[e] : (e - Ee);
  int cn = (e < Ee) ? ei[Ee + e] : (e - Ee);
  const float* hs = hbuf + (size_t)r * HC;
  float* ao = agg + (size_t)cn * HC;
#pragma unroll
  for (int h = 0; h < Hh; ++h) {
    float cf = alpha[(size_t)e * Hh + h] / (denom[cn * Hh + h] + 1e-16f);
    int b0 = h * Cc;
    atomAddF(&ao[b0 + lane],      cf * hs[b0 + lane]);
    atomAddF(&ao[b0 + 32 + lane], cf * hs[b0 + 32 + lane]);
  }
}

// ---------------- head mean + bias + relu ----------------
__global__ void headmean_kernel(const float* __restrict__ agg, const float* __restrict__ bias,
                                float* __restrict__ xout) {
  int id = blockIdx.x * 256 + threadIdx.x;
  if (id >= Nn * Cc) return;
  int n = id >> 6, c = id & 63;
  float acc = 0.0f;
#pragma unroll
  for (int h = 0; h < Hh; ++h) acc += agg[(size_t)n * HC + h * Cc + c];
  xout[id] = fmaxf(acc * 0.125f + bias[c], 0.0f);
}

// ---------------- BatchNorm stats (one block per channel) ----------------
__global__ void bn_stats_kernel(const float* __restrict__ y, float* __restrict__ meanOut,
                                float* __restrict__ rinvOut) {
  __shared__ float s1[256], s2[256];
  int c = blockIdx.x, t = threadIdx.x;
  float s = 0.0f, q = 0.0f;
  for (int n = t; n < Nn; n += 256) { float v = y[(size_t)n * Cc + c]; s += v; q += v * v; }
  s1[t] = s; s2[t] = q; __syncthreads();
  for (int k = 128; k > 0; k >>= 1) {
    if (t < k) { s1[t] += s1[t + k]; s2[t] += s2[t + k]; }
    __syncthreads();
  }
  if (t == 0) {
    float m = s1[0] / (float)Nn;
    float var = s2[0] / (float)Nn - m * m;
    meanOut[c] = m;
    rinvOut[c] = rsqrtf(var + 1e-5f);
  }
}

__global__ void bn_apply_kernel(const float* __restrict__ y, const float* __restrict__ mean,
                                const float* __restrict__ rinv, const float* __restrict__ g,
                                const float* __restrict__ beta, float* __restrict__ out) {
  int id = blockIdx.x * 256 + threadIdx.x;
  if (id >= Nn * Cc) return;
  int c = id & 63;
  out[id] = g[c] * (y[id] - mean[c]) * rinv[c] + beta[c];
}

// ---------------- heads: edge probs + pooled value ----------------
__global__ void edgeprob_kernel(const int* __restrict__ ei, const float* __restrict__ xfc,
                                const float* __restrict__ pW, const float* __restrict__ pb,
                                float* __restrict__ out) {
  int e = blockIdx.x * 256 + threadIdx.x;
  if (e >= Ee) return;
  int src = ei[e];
  const float* xp = xfc + (size_t)src * Cc;
  float acc = pb[0];
  for (int c = 0; c < Cc; ++c) acc += xp[c] * pW[c];
  out[e] = 1.0f / (1.0f + expf(-acc));
}

__global__ void pool_kernel(const float* __restrict__ xfc, const int* __restrict__ batch,
                            float* __restrict__ pooled, float* __restrict__ gcnt) {
  int id = blockIdx.x * 256 + threadIdx.x;
  if (id >= Nn * Cc) return;
  int n = id >> 6, c = id & 63;
  int g = batch[n];
  atomAddF(&pooled[g * Cc + c], xfc[id]);
  if (c == 0) atomAddF(&gcnt[g], 1.0f);
}

__global__ void value_kernel(const float* __restrict__ pooled, const float* __restrict__ gcnt,
                             const float* __restrict__ vW, const float* __restrict__ vb,
                             float* __restrict__ out) {
  int g = threadIdx.x;
  if (g >= NGg) return;
  float cnt = fmaxf(gcnt[g], 1.0f);
  float acc = 0.0f;
  for (int c = 0; c < Cc; ++c) acc += pooled[g * Cc + c] * vW[c];
  out[g] = tanhf(acc / cnt + vb[0]);
}

static inline int gridFor(long long n) { return (int)((n + 255) / 256); }

extern "C" void kernel_launch(void* const* d_in, const int* in_sizes, int n_in,
                              void* d_out, int out_size, void* d_ws, size_t ws_size,
                              hipStream_t stream) {
  (void)in_sizes; (void)n_in; (void)out_size; (void)ws_size;
  const float* x     = (const float*)d_in[0];
  const int*   ei    = (const int*)d_in[1];
  const float* ea    = (const float*)d_in[2];
  const int*   batch = (const int*)d_in[3];
  const float* W[3]  = {(const float*)d_in[4],  (const float*)d_in[10], (const float*)d_in[16]};
  const float* We[3] = {(const float*)d_in[5],  (const float*)d_in[11], (const float*)d_in[17]};
  const float* As[3] = {(const float*)d_in[6],  (const float*)d_in[12], (const float*)d_in[18]};
  const float* Ad[3] = {(const float*)d_in[7],  (const float*)d_in[13], (const float*)d_in[19]};
  const float* Ae[3] = {(const float*)d_in[8],  (const float*)d_in[14], (const float*)d_in[20]};
  const float* Bb[3] = {(const float*)d_in[9],  (const float*)d_in[15], (const float*)d_in[21]};
  const float* fc1W = (const float*)d_in[22]; const float* fc1b = (const float*)d_in[23];
  const float* fc1g = (const float*)d_in[24]; const float* fc1be = (const float*)d_in[25];
  const float* fc2W = (const float*)d_in[26]; const float* fc2b = (const float*)d_in[27];
  const float* fc2g = (const float*)d_in[28]; const float* fc2be = (const float*)d_in[29];
  const float* pW = (const float*)d_in[30]; const float* pb = (const float*)d_in[31];
  const float* vW = (const float*)d_in[32]; const float* vb = (const float*)d_in[33];
  float* out = (float*)d_out;

  char* base = (char*)d_ws;
  size_t off = 0;
  auto alloc = [&](size_t bytes) -> void* {
    void* p = base + off;
    off += (bytes + 255) & ~(size_t)255;
    return p;
  };
  float*    ea2    = (float*)alloc((size_t)E2T * FEd * 4);
  float*    lsum   = (float*)alloc((size_t)Nn * FEd * 4);
  float*    lcnt   = (float*)alloc((size_t)Nn * 4);
  float*    hbuf   = (float*)alloc((size_t)Nn * HC * 4);
  float*    agg    = (float*)alloc((size_t)Nn * HC * 4);
  float*    ssrc   = (float*)alloc((size_t)Nn * Hh * 4);
  float*    sdst   = (float*)alloc((size_t)Nn * Hh * 4);
  float*    alpha  = (float*)alloc((size_t)E2T * Hh * 4);
  unsigned* mmax   = (unsigned*)alloc((size_t)Nn * Hh * 4);
  float*    denom  = (float*)alloc((size_t)Nn * Hh * 4);
  float*    xa     = (float*)alloc((size_t)Nn * Cc * 4);
  float*    xb     = (float*)alloc((size_t)Nn * Cc * 4);
  float*    ybuf   = (float*)alloc((size_t)Nn * Cc * 4);
  _Float16* ah     = (_Float16*)alloc((size_t)Nn * Cc * 2);   // f16 activations
  _Float16* wt     = (_Float16*)alloc((size_t)HC * Cc * 2);
  _Float16* fwt    = (_Float16*)alloc((size_t)Cc * Cc * 2);
  float*    weatt  = (float*)alloc((size_t)FEd * Hh * 4);
  float*    bnmean = (float*)alloc((size_t)Cc * 4);
  float*    bnrinv = (float*)alloc((size_t)Cc * 4);
  float*    pooled = (float*)alloc((size_t)NGg * Cc * 4);
  float*    gcnt   = (float*)alloc((size_t)NGg * 4);

  // ---- Stage A: self-loop edge attrs ----
  hipMemsetAsync(lsum, 0, (size_t)Nn * FEd * 4, stream);
  hipMemsetAsync(lcnt, 0, (size_t)Nn * 4, stream);
  loop_sum_kernel<<<gridFor((long long)Ee * FEd), 256, 0, stream>>>(ei, ea, lsum, lcnt);
  build_ea2_kernel<<<gridFor((long long)E2T * FEd), 256, 0, stream>>>(ea, lsum, lcnt, ea2);

  // ---- Stage B: 3 GAT layers ----
  const float* xcur = x;
  float* xdst[3] = {xa, xb, xa};
  int fins[3] = {FNd, Cc, Cc};
  for (int l = 0; l < 3; ++l) {
    int K = fins[l];
    conv_transpose_f16<<<gridFor((long long)K * HC), 256, 0, stream>>>(W[l], wt, K, HC);
    we_att_kernel<<<1, 256, 0, stream>>>(We[l], Ae[l], weatt);
    conv_f16_kernel<<<gridFor((long long)Nn * K), 256, 0, stream>>>(xcur, ah, (long long)Nn * K);
    // h = x @ W  -> [N, 512]
    int tiles = (Nn / 16) * (HC / 64);
    gemm_wmma_f16_kernel<<<(tiles + 7) / 8, 256, 0, stream>>>(ah, wt, nullptr, hbuf,
                                                              Nn, HC, K, 0);
    scores_kernel<<<gridFor((long long)Nn * Hh), 256, 0, stream>>>(hbuf, As[l], Ad[l], ssrc, sdst);
    hipMemsetAsync(mmax, 0, (size_t)Nn * Hh * 4, stream);   // fkey order: 0 < key(any real)
    hipMemsetAsync(denom, 0, (size_t)Nn * Hh * 4, stream);
    hipMemsetAsync(agg, 0, (size_t)Nn * HC * 4, stream);
    alpha1_kernel<<<gridFor((long long)E2T * Hh), 256, 0, stream>>>(ei, ea2, weatt, ssrc, sdst,
                                                                    alpha, mmax);
    alpha2_kernel<<<gridFor((long long)E2T * Hh), 256, 0, stream>>>(ei, alpha, mmax, denom);
    agg_kernel<<<(E2T + 7) / 8, 256, 0, stream>>>(ei, hbuf, alpha, denom, agg);
    headmean_kernel<<<gridFor((long long)Nn * Cc), 256, 0, stream>>>(agg, Bb[l], xdst[l]);
    xcur = xdst[l];
  }

  // ---- Stage C: fc1 -> BN, fc2 -> BN ----
  int ftiles = (Nn / 16) * (Cc / 64);
  conv_transpose_f16<<<gridFor((long long)Cc * Cc), 256, 0, stream>>>(fc1W, fwt, Cc, Cc);
  conv_f16_kernel<<<gridFor((long long)Nn * Cc), 256, 0, stream>>>(xa, ah, (long long)Nn * Cc);
  gemm_wmma_f16_kernel<<<(ftiles + 7) / 8, 256, 0, stream>>>(ah, fwt, fc1b, ybuf, Nn, Cc, Cc, 1);
  bn_stats_kernel<<<Cc, 256, 0, stream>>>(ybuf, bnmean, bnrinv);
  bn_apply_kernel<<<gridFor((long long)Nn * Cc), 256, 0, stream>>>(ybuf, bnmean, bnrinv,
                                                                   fc1g, fc1be, xb);
  conv_transpose_f16<<<gridFor((long long)Cc * Cc), 256, 0, stream>>>(fc2W, fwt, Cc, Cc);
  conv_f16_kernel<<<gridFor((long long)Nn * Cc), 256, 0, stream>>>(xb, ah, (long long)Nn * Cc);
  gemm_wmma_f16_kernel<<<(ftiles + 7) / 8, 256, 0, stream>>>(ah, fwt, fc2b, ybuf, Nn, Cc, Cc, 1);
  bn_stats_kernel<<<Cc, 256, 0, stream>>>(ybuf, bnmean, bnrinv);
  bn_apply_kernel<<<gridFor((long long)Nn * Cc), 256, 0, stream>>>(ybuf, bnmean, bnrinv,
                                                                   fc2g, fc2be, xa);

  // ---- Stage D: heads ----
  edgeprob_kernel<<<gridFor((long long)Ee), 256, 0, stream>>>(ei, xa, pW, pb, out);
  hipMemsetAsync(pooled, 0, (size_t)NGg * Cc * 4, stream);
  hipMemsetAsync(gcnt, 0, (size_t)NGg * 4, stream);
  pool_kernel<<<gridFor((long long)Nn * Cc), 256, 0, stream>>>(xa, batch, pooled, gcnt);
  value_kernel<<<1, 64, 0, stream>>>(pooled, gcnt, vW, vb, out + Ee);
}
